// PINNsformer_68298569941018
// MI455X (gfx1250) — matrix operand order repaired and checked by reading, not compiled, not run on hardware
//
#include <hip/hip_runtime.h>
#include <math.h>
#include <stdint.h>

// ---------------- types for WMMA / TDM ----------------
typedef __attribute__((ext_vector_type(16))) __bf16      v16bf;
typedef __attribute__((ext_vector_type(8)))  float       v8f;
typedef __attribute__((ext_vector_type(4)))  unsigned int uint4v;
typedef __attribute__((ext_vector_type(4)))  unsigned int u32x4;
typedef __attribute__((ext_vector_type(8)))  int          i32x8;
typedef __attribute__((ext_vector_type(4)))  int          i32x4;
typedef unsigned short ushort;

union Frag16 {
    v16bf  v;
    ushort u[16];
    uint4v q[2];
};

__device__ __forceinline__ ushort f2bf(float f) {
    unsigned u = __float_as_uint(f);
    unsigned r = u + 0x7FFFu + ((u >> 16) & 1u);   // round-to-nearest-even
    return (ushort)(r >> 16);
}
__device__ __forceinline__ float bf2f(ushort h) {
    return __uint_as_float(((unsigned)h) << 16);
}

// problem constants
#define Bb   256
#define Ss   256
#define Ll   4
#define Dd   512
#define Hh   8
#define HDd  64
#define HVv  128
#define Vv   1024
#define FFNn 128
#define DHh  512
#define Mm   (Bb * Ss)   // 65536

// ---------------- Tensor Data Mover (CDNA5) ----------------
#if defined(__has_builtin)
# if __has_builtin(__builtin_amdgcn_tensor_load_to_lds) && __has_builtin(__builtin_amdgcn_s_wait_tensorcnt)
#  define HAVE_TDM 1
# endif
#endif

#ifdef HAVE_TDM
// 2D tile load: tile_d0 contiguous elements (2B each) x tile_d1 rows,
// row stride stride0 elements; lands contiguously (row-major) in LDS.
__device__ __forceinline__ void tdm_load_2d(unsigned lds_addr, const void* gaddr,
                                            unsigned tensor_d0, unsigned tensor_d1,
                                            unsigned tile_d0, unsigned tile_d1,
                                            unsigned long long stride0)
{
    unsigned long long ga = (unsigned long long)(uintptr_t)gaddr;
    u32x4 g0;
    g0[0] = 1u;                                                 // count=1 (valid user D#)
    g0[1] = lds_addr;                                           // LDS byte address
    g0[2] = (unsigned)(ga & 0xFFFFFFFFu);                       // global_addr[31:0]
    g0[3] = (unsigned)((ga >> 32) & 0x1FFFFFFu) | (2u << 30);   // addr[56:32] | type=2
    i32x8 g1;
    g1[0] = (int)(1u << 16);                                    // data_size=1 (2 bytes)
    g1[1] = (int)((tensor_d0 & 0xFFFFu) << 16);                 // tensor_dim0[15:0]
    g1[2] = (int)((tensor_d0 >> 16) | ((tensor_d1 & 0xFFFFu) << 16));
    g1[3] = (int)((tensor_d1 >> 16) | (tile_d0 << 16));         // tile_dim0
    g1[4] = (int)tile_d1;                                       // tile_dim1 (tile_dim2=0)
    g1[5] = (int)(unsigned)(stride0 & 0xFFFFFFFFull);           // dim0_stride[31:0]
    g1[6] = (int)(unsigned)((stride0 >> 32) & 0xFFFFull);       // dim0_stride[47:32]
    g1[7] = 0;
    i32x4 z = {};
#if defined(__clang_major__) && (__clang_major__ >= 23)
    i32x8 z8 = {};
    __builtin_amdgcn_tensor_load_to_lds(g0, g1, z, z, z8, 0);
#else
    __builtin_amdgcn_tensor_load_to_lds(g0, g1, z, z, 0);
#endif
}
#endif

// =====================================================================
// Generic bf16 WMMA GEMM:  C(f32 or bf16)[M,N] = A_bf16[M,K] * B_bf16[K,N]
//                          (+bias[N]) (+resid[M,N])
// Block tile 128x64, 8 waves of 32x32, K-step 32.
// A tile staged by the Tensor Data Mover when available.
// =====================================================================
#define BM 128
#define BN 64
#define BK 32

__global__ __launch_bounds__(256)
void k_gemm_bf16(const ushort* __restrict__ A, const ushort* __restrict__ B,
                 int M, int N, int K,
                 const float* __restrict__ bias, const float* __restrict__ resid,
                 float* __restrict__ Cf, ushort* __restrict__ Cbf)
{
    __shared__ __align__(16) ushort Alds[BM * BK];   // [m][k]       8KB
    __shared__ __align__(16) ushort Blds[BN * BK];   // [n][k] (T)   4KB

    const int tid  = threadIdx.x;
    const int wave = tid >> 5;
    const int lane = tid & 31;
    const int hlf  = lane >> 4;
    const int l15  = lane & 15;

    const int nBlk = N / BN;
    const int bm   = (int)(blockIdx.x / nBlk) * BM;
    const int bn   = (int)(blockIdx.x % nBlk) * BN;

    const int wm = (wave >> 1) * 32;   // 4 waves down M
    const int wn = (wave & 1) * 32;    // 2 waves across N

    v8f acc[2][2] = {};

    for (int kt = 0; kt < K; kt += BK) {
#ifdef HAVE_TDM
        // --- stage A tile via Tensor Data Mover: 128 rows x 32 elems ---
        if (wave == 0) {
            tdm_load_2d((unsigned)(uintptr_t)(void*)Alds,
                        &A[(size_t)bm * K + kt],
                        (unsigned)K, (unsigned)M,
                        (unsigned)BK, (unsigned)BM,
                        (unsigned long long)K);
        }
#else
        // --- stage A tile: 128x32 bf16, 16B chunks, 2 per thread ---
        #pragma unroll
        for (int i = 0; i < 2; ++i) {
            int c = tid * 2 + i;
            int r = c >> 2, kc = (c & 3) * 8;
            *(uint4v*)&Alds[r * BK + kc] =
                *(const uint4v*)&A[(size_t)(bm + r) * K + kt + kc];
        }
#endif
        // --- stage B tile transposed: B[kt+k][bn+n] -> Blds[n*BK + k] ---
        {
            int r  = tid >> 3;        // k row 0..31
            int nc = (tid & 7) * 8;   // n col base
            const ushort* src = &B[(size_t)(kt + r) * N + bn + nc];
            #pragma unroll
            for (int j = 0; j < 8; ++j) Blds[(nc + j) * BK + r] = src[j];
        }
#ifdef HAVE_TDM
        if (wave == 0) __builtin_amdgcn_s_wait_tensorcnt(0);
#endif
        __syncthreads();

        if (kt + BK < K) {  // CDNA5 global_prefetch_b8 for next tiles
            __builtin_prefetch(&A[(size_t)(bm + (tid >> 1)) * K + kt + BK], 0, 3);
            __builtin_prefetch(&B[(size_t)(kt + BK + (tid & 31)) * N + bn], 0, 3);
        }

        #pragma unroll
        for (int mi = 0; mi < 2; ++mi) {
            Frag16 a;
            int m = wm + mi * 16 + l15;
            a.q[0] = *(const uint4v*)&Alds[m * BK + hlf * 8];
            a.q[1] = *(const uint4v*)&Alds[m * BK + 16 + hlf * 8];
            #pragma unroll
            for (int ni = 0; ni < 2; ++ni) {
                Frag16 b;
                int n = wn + ni * 16 + l15;
                b.q[0] = *(const uint4v*)&Blds[n * BK + hlf * 8];
                b.q[1] = *(const uint4v*)&Blds[n * BK + 16 + hlf * 8];
                acc[mi][ni] = __builtin_amdgcn_wmma_f32_16x16x32_bf16(
                    false, a.v, false, b.v, (short)0, acc[mi][ni], false, false);
            }
        }
        __syncthreads();
    }

    // ----- epilogue -----
    #pragma unroll
    for (int mi = 0; mi < 2; ++mi)
    #pragma unroll
    for (int ni = 0; ni < 2; ++ni) {
        int col = bn + wn + ni * 16 + l15;
        float bv = bias ? bias[col] : 0.f;
        #pragma unroll
        for (int j = 0; j < 8; ++j) {
            int row = bm + wm + mi * 16 + hlf * 8 + j;
            float v = acc[mi][ni][j] + bv;
            if (resid) v += resid[(size_t)row * N + col];
            if (Cf)  Cf[(size_t)row * N + col]  = v;
            if (Cbf) Cbf[(size_t)row * N + col] = f2bf(v);
        }
    }
}

// =====================================================================
// Retention attention: per (b,h,128-row strip).
//   A = (Q xpos) (K xpos)^T * Dm ;  Y = A V
// Q/K bf16 [b,h,s,64]; V stored pre-transposed bf16 [b,h,v,s].
// Each of 8 waves owns a 16-row tile. 64KB LDS for A strips.
// =====================================================================
__global__ __launch_bounds__(256)
void k_attention(const ushort* __restrict__ Qbf, const ushort* __restrict__ Kbf,
                 const ushort* __restrict__ Vt,  float* __restrict__ Y)
{
    __shared__ __align__(16) ushort As[8 * 16 * 256];   // 64KB, per-wave strips

    const int tid  = threadIdx.x;
    const int wave = tid >> 5;
    const int lane = tid & 31;
    const int hlf  = lane >> 4;
    const int l15  = lane & 15;

    const int blk   = blockIdx.x;
    const int strip = (blk & 1) * 128;       // S/128 = 2 strips
    const int bh    = blk >> 1;
    const int hh    = bh & (Hh - 1);
    const int bb    = bh >> 3;

    const size_t qoff = (size_t)(bb * Hh + hh) * Ss * HDd;   // Q/K base
    const size_t voff = (size_t)(bb * Hh + hh) * HVv * Ss;   // V^T base

    // per-head decay gamma
    const float lg0 = -3.46573590f, lg1 = -6.23832463f;      // ln(1/32), ln(1/512)
    const float gam = 1.f - __expf(lg0 + (lg1 - lg0) * (float)hh / 7.f);
    const float l2g = __log2f(gam);

    const int srow = strip + wave * 16;      // this wave's 16 rows

    // Q fragments for this wave (16x64, two 32-wide k chunks) — direct global
    Frag16 aq[2];
    #pragma unroll
    for (int kk = 0; kk < 2; ++kk) {
        size_t base = qoff + (size_t)(srow + l15) * HDd + kk * 32;
        aq[kk].q[0] = *(const uint4v*)&Qbf[base + hlf * 8];
        aq[kk].q[1] = *(const uint4v*)&Qbf[base + 16 + hlf * 8];
    }

    ushort* myA = &As[wave * 16 * 256];

    // ---- phase 1: A = Q K^T, apply decay, stash bf16 strip in LDS ----
    for (int nt = 0; nt < 16; ++nt) {
        v8f a = {};
        #pragma unroll
        for (int kk = 0; kk < 2; ++kk) {
            Frag16 b;   // B = K^T : table order, u[0..7] = K[n][kk*32 + hlf*8 + i], u[8..15] = K[n][kk*32 + 16 + hlf*8 + i]
            size_t base = qoff + (size_t)(nt * 16 + l15) * HDd + kk * 32;
            b.q[0] = *(const uint4v*)&Kbf[base + hlf * 8];
            b.q[1] = *(const uint4v*)&Kbf[base + 16 + hlf * 8];
            a = __builtin_amdgcn_wmma_f32_16x16x32_bf16(
                false, aq[kk].v, false, b.v, (short)0, a, false, false);
        }
        const int mcol = nt * 16 + l15;      // absolute key position
        #pragma unroll
        for (int j = 0; j < 8; ++j) {
            int r = srow + hlf * 8 + j;      // absolute query position
            int d = r - mcol;
            float f = (d >= 0) ? exp2f((float)d * l2g) : 0.f;
            myA[(hlf * 8 + j) * 256 + mcol] = f2bf(a[j] * f);
        }
    }

    // ---- phase 2: Y = A V  (K-dim = 256, 8 steps of 32) ----
    v8f yacc[8] = {};
    for (int kt = 0; kt < 8; ++kt) {
        Frag16 a;
        int base = l15 * 256 + kt * 32;
        a.q[0] = *(const uint4v*)&myA[base + hlf * 8];
        a.q[1] = *(const uint4v*)&myA[base + 16 + hlf * 8];
        #pragma unroll
        for (int nv = 0; nv < 8; ++nv) {
            Frag16 b;   // table order: u[0..7] = V[kt*32 + hlf*8 + i][n], u[8..15] = V[kt*32 + 16 + hlf*8 + i][n] (from Vt[n*256 + ...])
            size_t vb = voff + (size_t)(nv * 16 + l15) * Ss + kt * 32;
            b.q[0] = *(const uint4v*)&Vt[vb + hlf * 8];
            b.q[1] = *(const uint4v*)&Vt[vb + 16 + hlf * 8];
            yacc[nv] = __builtin_amdgcn_wmma_f32_16x16x32_bf16(
                false, a.v, false, b.v, (short)0, yacc[nv], false, false);
        }
    }

    // ---- write Y[b,s,h,v] f32 ----
    #pragma unroll
    for (int nv = 0; nv < 8; ++nv) {
        int v = nv * 16 + l15;
        #pragma unroll
        for (int j = 0; j < 8; ++j) {
            int s = srow + hlf * 8 + j;
            Y[(((size_t)bb * Ss + s) * Hh + hh) * HVv + v] = yacc[nv][j];
        }
    }
}

// =====================================================================
// elementwise / reduction kernels
// =====================================================================
__global__ __launch_bounds__(256)
void k_embed(const float* __restrict__ x, const float* __restrict__ t,
             const float* __restrict__ W, const float* __restrict__ b,
             float* __restrict__ X)
{
    size_t i = (size_t)blockIdx.x * 256 + threadIdx.x;   // M*D
    int d = (int)(i & (Dd - 1));
    size_t m = i >> 9;
    X[i] = x[m] * W[d] + t[m] * W[Dd + d] + b[d];
}

// wave-per-row LayerNorm over D=512, bf16 out
__global__ __launch_bounds__(256)
void k_layernorm_bf16(const float* __restrict__ X, const float* __restrict__ w,
                      const float* __restrict__ b, ushort* __restrict__ out)
{
    int row  = blockIdx.x * 8 + (threadIdx.x >> 5);
    int lane = threadIdx.x & 31;
    const float* xr = X + (size_t)row * Dd;
    float s = 0.f, ss = 0.f;
    for (int i = lane; i < Dd; i += 32) { float v = xr[i]; s += v; ss += v * v; }
    for (int m = 16; m; m >>= 1) { s += __shfl_xor(s, m, 32); ss += __shfl_xor(ss, m, 32); }
    float mu = s / Dd, var = ss / Dd - mu * mu;
    float r = rsqrtf(var + 1e-5f);
    for (int i = lane; i < Dd; i += 32)
        out[(size_t)row * Dd + i] = f2bf((xr[i] - mu) * r * w[i] + b[i]);
}

// xPos rotary on Q,K + V transpose; reads QKVG f32 [M,3072]
__global__ __launch_bounds__(256)
void k_xpos(const float* __restrict__ QKVG, ushort* __restrict__ Qbf,
            ushort* __restrict__ Kbf, ushort* __restrict__ Vt)
{
    size_t tI = (size_t)blockIdx.x * 256 + threadIdx.x;  // B*H*S*64
    int e  = (int)(tI & 63);
    int s  = (int)((tI >> 6) & 255);
    int hh = (int)((tI >> 14) & 7);
    int b  = (int)(tI >> 17);
    size_t m = (size_t)b * Ss + s;
    const float* row = QKVG + m * 3072;

    float q  = row[hh * 64 + e],       k  = row[512 + hh * 64 + e];
    int   ep = e ^ 1;
    float qp = row[hh * 64 + ep],      kp = row[512 + hh * 64 + ep];

    int   i    = e >> 1;
    float sv   = (2.f * i + 0.4f * 64.f) / (1.4f * 64.f);
    float sc   = __powf(sv, (float)s / 512.f);
    float invf = __powf(10000.f, -(float)i / 32.f);
    float ang  = (float)s * invf;
    float sn = __sinf(ang), cs = __cosf(ang);
    float rotq = (e & 1) ? qp : -qp;
    float rotk = (e & 1) ? kp : -kp;
    float qo = (q * cs + rotq * sn) * sc;
    float ko = (k * cs + rotk * sn) / sc;

    size_t o = (((size_t)b * Hh + hh) * Ss + s) * HDd + e;
    Qbf[o] = f2bf(qo);
    Kbf[o] = f2bf(ko);

    // V: copy two elements into V^T [b,h,v,s]
    float v0 = row[1024 + hh * 128 + 2 * e];
    float v1 = row[1024 + hh * 128 + 2 * e + 1];
    size_t ov = (((size_t)b * Hh + hh) * HVv + 2 * e) * Ss + s;
    Vt[ov]      = f2bf(v0);
    Vt[ov + Ss] = f2bf(v1);
}

// group-norm over HV=128 + SiLU gate from G; bf16 out [M,V]
__global__ __launch_bounds__(256)
void k_gn_gate(const float* __restrict__ Y, const float* __restrict__ QKVG,
               const float* __restrict__ gw, const float* __restrict__ gb,
               ushort* __restrict__ GY)
{
    int t = blockIdx.x * 256 + threadIdx.x;   // one per (b,s,h)
    int hh = t & 7;
    size_t m = (size_t)(t >> 3);
    const float* y = Y + m * Vv + hh * HVv;
    float s = 0.f, ss = 0.f;
    for (int i = 0; i < HVv; ++i) { float v = y[i]; s += v; ss += v * v; }
    float mu = s / HVv, var = ss / HVv - mu * mu;
    float r = rsqrtf(var + 1e-5f);
    const float* g = QKVG + m * 3072 + 2048 + hh * 128;
    for (int i = 0; i < HVv; ++i) {
        float yn = (y[i] - mu) * r * gw[hh * 128 + i] + gb[hh * 128 + i];
        float gv = g[i];
        float gate = gv / (1.f + __expf(-gv));
        GY[m * Vv + hh * HVv + i] = f2bf(gate * yn);
    }
}

__global__ __launch_bounds__(256)
void k_gelu_bf16(const float* __restrict__ in, ushort* __restrict__ out)
{
    size_t i = (size_t)blockIdx.x * 256 + threadIdx.x;
    float x = in[i];
    out[i] = f2bf(0.5f * x * (1.f + erff(x * 0.70710678f)));
}

__global__ __launch_bounds__(256)
void k_wave_bf16(const float* __restrict__ in, const float* __restrict__ wave_w,
                 int rowsel, ushort* __restrict__ out)
{
    size_t i = (size_t)blockIdx.x * 256 + threadIdx.x;
    float w0 = wave_w[rowsel * 2], w1 = wave_w[rowsel * 2 + 1];
    float h = in[i];
    out[i] = f2bf(w0 * __sinf(h) + w1 * __cosf(h));
}

__global__ __launch_bounds__(256)
void k_f32_to_bf16(const float* __restrict__ in, ushort* __restrict__ out)
{
    size_t i = (size_t)blockIdx.x * 256 + threadIdx.x;
    out[i] = f2bf(in[i]);
}

// final projection DH -> 1 : wave-per-row dot product
__global__ __launch_bounds__(256)
void k_head3(const ushort* __restrict__ Hbf, const float* __restrict__ W,
             const float* __restrict__ b, float* __restrict__ out)
{
    int row  = blockIdx.x * 8 + (threadIdx.x >> 5);
    int lane = threadIdx.x & 31;
    float s = 0.f;
    for (int i = lane; i < DHh; i += 32) s += bf2f(Hbf[(size_t)row * DHh + i]) * W[i];
    for (int m = 16; m; m >>= 1) s += __shfl_xor(s, m, 32);
    if (lane == 0) out[row] = s + b[0];
}

// =====================================================================
// weight conversion
// =====================================================================
__global__ __launch_bounds__(256)
void k_conv_qkvg(const float* __restrict__ WQ, const float* __restrict__ WK,
                 const float* __restrict__ WV, const float* __restrict__ WG,
                 ushort* __restrict__ Wcat)
{
    size_t t = (size_t)blockIdx.x * 256 + threadIdx.x;   // L*512*3072
    int n = (int)(t % 3072);
    size_t r = t / 3072;
    int k = (int)(r & (Dd - 1));
    int l = (int)(r >> 9);
    float v;
    if (n < 512) {
        int h = n >> 6, e = n & 63;
        v = WQ[(((size_t)l * Hh + h) * Dd + k) * HDd + e];
    } else if (n < 1024) {
        int n1 = n - 512, h = n1 >> 6, e = n1 & 63;
        v = WK[(((size_t)l * Hh + h) * Dd + k) * HDd + e];
    } else if (n < 2048) {
        int n2 = n - 1024, h = n2 >> 7, e = n2 & 127;
        v = WV[(((size_t)l * Hh + h) * Dd + k) * HVv + e];
    } else {
        int n3 = n - 2048;
        v = WG[((size_t)l * Dd + k) * Vv + n3];
    }
    Wcat[t] = f2bf(v);
}

__global__ __launch_bounds__(256)
void k_conv_mat(const float* __restrict__ src, ushort* __restrict__ dst)
{
    size_t i = (size_t)blockIdx.x * 256 + threadIdx.x;
    dst[i] = f2bf(src[i]);
}

// =====================================================================
// host launcher
// =====================================================================
extern "C" void kernel_launch(void* const* d_in, const int* in_sizes, int n_in,
                              void* d_out, int out_size, void* d_ws, size_t ws_size,
                              hipStream_t stream)
{
    (void)in_sizes; (void)n_in; (void)out_size; (void)ws_size;

    const float* x      = (const float*)d_in[0];
    const float* t      = (const float*)d_in[1];
    const float* emb_W  = (const float*)d_in[2];
    const float* emb_b  = (const float*)d_in[3];
    const float* ln1_w  = (const float*)d_in[4];
    const float* ln1_b  = (const float*)d_in[5];
    const float* ln2_w  = (const float*)d_in[6];
    const float* ln2_b  = (const float*)d_in[7];
    const float* WQ     = (const float*)d_in[8];
    const float* WK     = (const float*)d_in[9];
    const float* WVp    = (const float*)d_in[10];
    const float* WG     = (const float*)d_in[11];
    const float* WO     = (const float*)d_in[12];
    const float* gn_w   = (const float*)d_in[13];
    const float* gn_b   = (const float*)d_in[14];
    const float* f1_W   = (const float*)d_in[15];
    const float* f1_b   = (const float*)d_in[16];
    const float* f2_W   = (const float*)d_in[17];
    const float* f2_b   = (const float*)d_in[18];
    const float* h1_W   = (const float*)d_in[19];
    const float* h1_b   = (const float*)d_in[20];
    const float* wave_w = (const float*)d_in[21];
    const float* h2_W   = (const float*)d_in[22];
    const float* h2_b   = (const float*)d_in[23];
    const float* h3_W   = (const float*)d_in[24];
    const float* h3_b   = (const float*)d_in[25];
    float* out = (float*)d_out;

    // bump allocator over d_ws
    size_t off = 0;
    auto alloc = [&](size_t bytes) -> void* {
        void* p = (char*)d_ws + off;
        off = (off + bytes + 255) & ~(size_t)255;
        return p;
    };

    ushort* Wcat = (ushort*)alloc((size_t)Ll * Dd * 3072 * 2);
    ushort* WObf = (ushort*)alloc((size_t)Ll * Vv * Dd * 2);
    ushort* F1w  = (ushort*)alloc((size_t)Ll * Dd * FFNn * 2);
    ushort* F2w  = (ushort*)alloc((size_t)Ll * FFNn * Dd * 2);
    ushort* H1w  = (ushort*)alloc((size_t)Dd * DHh * 2);
    ushort* H2w  = (ushort*)alloc((size_t)DHh * DHh * 2);

    float*  X    = (float*) alloc((size_t)Mm * Dd * 4);
    float*  Xr   = (float*) alloc((size_t)Mm * Dd * 4);
    ushort* Xnbf = (ushort*)alloc((size_t)Mm * Dd * 2);
    float*  QKVG = (float*) alloc((size_t)Mm * 3072 * 4);
    ushort* Qbf  = (ushort*)alloc((size_t)Mm * Dd * 2);
    ushort* Kbf  = (ushort*)alloc((size_t)Mm * Dd * 2);
    ushort* Vt   = (ushort*)alloc((size_t)Mm * Vv * 2);
    float*  Ybuf = (float*) alloc((size_t)Mm * Vv * 4);
    ushort* GYbf = (ushort*)alloc((size_t)Mm * Vv * 2);
    float*  F1   = (float*) alloc((size_t)Mm * FFNn * 4);
    ushort* F1bf = (ushort*)alloc((size_t)Mm * FFNn * 2);
    float*  Hf   = (float*) alloc((size_t)Mm * DHh * 4);
    ushort* Hbf  = (ushort*)alloc((size_t)Mm * DHh * 2);

    // ---- weight conversion (deterministic each call) ----
    k_conv_qkvg<<<(Ll * Dd * 3072) / 256, 256, 0, stream>>>(WQ, WK, WVp, WG, Wcat);
    k_conv_mat<<<(Ll * Vv * Dd) / 256, 256, 0, stream>>>(WO, WObf);
    k_conv_mat<<<(Ll * Dd * FFNn) / 256, 256, 0, stream>>>(f1_W, F1w);
    k_conv_mat<<<(Ll * FFNn * Dd) / 256, 256, 0, stream>>>(f2_W, F2w);
    k_conv_mat<<<(Dd * DHh) / 256, 256, 0, stream>>>(h1_W, H1w);
    k_conv_mat<<<(DHh * DHh) / 256, 256, 0, stream>>>(h2_W, H2w);

    // ---- embedding ----
    k_embed<<<(Mm * Dd) / 256, 256, 0, stream>>>(x, t, emb_W, emb_b, X);

    // ---- transformer layers ----
    for (int l = 0; l < Ll; ++l) {
        // LN1 -> bf16
        k_layernorm_bf16<<<Mm / 8, 256, 0, stream>>>(X, ln1_w + l * Dd, ln1_b + l * Dd, Xnbf);

        // fused QKV+G projection: [M,512] x [512,3072] -> f32
        k_gemm_bf16<<<(Mm / BM) * (3072 / BN), 256, 0, stream>>>(
            Xnbf, Wcat + (size_t)l * Dd * 3072, Mm, 3072, Dd,
            nullptr, nullptr, QKVG, nullptr);

        // xPos rotary + V transpose (one thread per (b,h,s,e))
        k_xpos<<<(Mm * Dd) / 256, 256, 0, stream>>>(QKVG, Qbf, Kbf, Vt);

        // retention attention (WMMA)
        k_attention<<<Bb * Hh * (Ss / 128), 256, 0, stream>>>(Qbf, Kbf, Vt, Ybuf);

        // group-norm + SiLU gate -> bf16 [M,V]
        k_gn_gate<<<(Mm * Hh) / 256, 256, 0, stream>>>(Ybuf, QKVG,
                                                       gn_w + l * Vv, gn_b + l * Vv, GYbf);

        // output projection + residual: [M,1024] x [1024,512] + X -> Xr
        k_gemm_bf16<<<(Mm / BM) * (Dd / BN), 256, 0, stream>>>(
            GYbf, WObf + (size_t)l * Vv * Dd, Mm, Dd, Vv,
            nullptr, X, Xr, nullptr);

        // LN2 -> bf16
        k_layernorm_bf16<<<Mm / 8, 256, 0, stream>>>(Xr, ln2_w + l * Dd, ln2_b + l * Dd, Xnbf);

        // FFN up: [M,512] x [512,128] + b -> f32
        k_gemm_bf16<<<(Mm / BM) * (FFNn / BN), 256, 0, stream>>>(
            Xnbf, F1w + (size_t)l * Dd * FFNn, Mm, FFNn, Dd,
            f1_b + l * FFNn, nullptr, F1, nullptr);

        // GELU -> bf16
        k_gelu_bf16<<<(Mm * FFNn) / 256, 256, 0, stream>>>(F1, F1bf);

        // FFN down + bias + residual(Xr) -> X
        k_gemm_bf16<<<(Mm / BM) * (Dd / BN), 256, 0, stream>>>(
            F1bf, F2w + (size_t)l * FFNn * Dd, Mm, Dd, FFNn,
            f2_b + l * Dd, Xr, X, nullptr);
    }

    // ---- head MLP ----
    k_f32_to_bf16<<<(Mm * Dd) / 256, 256, 0, stream>>>(X, Xnbf);

    k_gemm_bf16<<<(Mm / BM) * (DHh / BN), 256, 0, stream>>>(
        Xnbf, H1w, Mm, DHh, Dd, h1_b, nullptr, Hf, nullptr);
    k_wave_bf16<<<(Mm * DHh) / 256, 256, 0, stream>>>(Hf, wave_w, 0, Hbf);

    k_gemm_bf16<<<(Mm / BM) * (DHh / BN), 256, 0, stream>>>(
        Hbf, H2w, Mm, DHh, DHh, h2_b, nullptr, Hf, nullptr);
    k_wave_bf16<<<(Mm * DHh) / 256, 256, 0, stream>>>(Hf, wave_w, 1, Hbf);

    k_head3<<<Mm / 8, 256, 0, stream>>>(Hbf, h3_W, h3_b, out);
}
